// EquiRNN_31568009626353
// MI455X (gfx1250) — matrix operand order repaired and checked by reading, not compiled, not run on hardware
//
#include <hip/hip_runtime.h>

// ---- problem constants (match reference) ----
#define GG 24
#define TT 2048
#define VV 2048
#define KK 512

typedef _Float16 h8   __attribute__((ext_vector_type(8)));
typedef _Float16 v16h __attribute__((ext_vector_type(16)));
typedef float    v8f  __attribute__((ext_vector_type(8)));

#define SHUF16(lo, hi) __builtin_shufflevector(lo, hi, 0,1,2,3,4,5,6,7,8,9,10,11,12,13,14,15)

// ---------- 1) recover tokens from one-hot rows ----------
__global__ void eqr_tok_kernel(const float* __restrict__ seq, int* __restrict__ tok) {
    int t = blockIdx.x;
    for (int v = threadIdx.x; v < VV; v += blockDim.x) {
        if (seq[(size_t)t * VV + v] > 0.5f) tok[t] = v;
    }
}

// ---------- 2) f32 -> f16 conversion (optionally transposed) ----------
__global__ void eqr_cvt_kernel(const float* __restrict__ src, _Float16* __restrict__ dst,
                               int R, int C, int trans) {
    int i = blockIdx.x * blockDim.x + threadIdx.x;
    if (i >= R * C) return;
    int r = i / C, c = i - r * C;
    if (trans) dst[(size_t)c * R + r] = (_Float16)src[i];
    else       dst[i]                 = (_Float16)src[i];
}

// ---------- 3) xproj GEMM: (T*G, K) x (K, K) + b, WMMA f16->f32 ----------
// One wave per 16(M)x64(N) strip: A fragment loaded once per k-step and
// reused across 4 WMMAs (4 accumulators), cutting A-side L2 traffic 4x.
__global__ void __launch_bounds__(32)
eqr_xproj_kernel(const _Float16* __restrict__ We16,   // (V, K) row-major f16
                 const _Float16* __restrict__ WxT,    // (K, K) transposed: [n][k]
                 const int* __restrict__ perms,       // (G, V)
                 const int* __restrict__ tok,         // (T)
                 const float* __restrict__ bias,      // (K)
                 _Float16* __restrict__ xproj) {      // (T*G, K) f16
    int mt   = blockIdx.x >> 3;   // 0..3071  (49152/16 M-tiles)
    int ng   = blockIdx.x & 7;    // 0..7     (8 groups of 4 N-tiles)
    int lane = threadIdx.x;
    int hi   = lane >> 4;
    int nl   = lane & 15;

    // A row for this lane (rows of the M-tile are indexed by lane&15)
    int m = mt * 16 + nl;
    int t = m / GG;
    int g = m - t * GG;
    int row = perms[g * VV + tok[t]];
    const _Float16* arow = We16 + (size_t)row * KK;

    const int n0 = ng * 64 + nl;  // column of N-tile 0 for this lane
    const _Float16* bcol0 = WxT + (size_t)(n0 +  0) * KK + hi * 16;
    const _Float16* bcol1 = WxT + (size_t)(n0 + 16) * KK + hi * 16;
    const _Float16* bcol2 = WxT + (size_t)(n0 + 32) * KK + hi * 16;
    const _Float16* bcol3 = WxT + (size_t)(n0 + 48) * KK + hi * 16;

    v8f c0 = {}, c1 = {}, c2 = {}, c3 = {};
    for (int k0 = 0; k0 < KK; k0 += 32) {
        h8 a0 = *(const h8*)(arow + k0 + hi * 16);
        h8 a1 = *(const h8*)(arow + k0 + hi * 16 + 8);
        v16h a = SHUF16(a0, a1);

        h8 p0 = *(const h8*)(bcol0 + k0); h8 p1 = *(const h8*)(bcol0 + k0 + 8);
        v16h b0 = SHUF16(p0, p1);
        c0 = __builtin_amdgcn_wmma_f32_16x16x32_f16(false, a, false, b0, (short)0, c0, false, false);

        h8 q0 = *(const h8*)(bcol1 + k0); h8 q1 = *(const h8*)(bcol1 + k0 + 8);
        v16h b1 = SHUF16(q0, q1);
        c1 = __builtin_amdgcn_wmma_f32_16x16x32_f16(false, a, false, b1, (short)0, c1, false, false);

        h8 r0 = *(const h8*)(bcol2 + k0); h8 r1 = *(const h8*)(bcol2 + k0 + 8);
        v16h b2 = SHUF16(r0, r1);
        c2 = __builtin_amdgcn_wmma_f32_16x16x32_f16(false, a, false, b2, (short)0, c2, false, false);

        h8 s0 = *(const h8*)(bcol3 + k0); h8 s1 = *(const h8*)(bcol3 + k0 + 8);
        v16h b3 = SHUF16(s0, s1);
        c3 = __builtin_amdgcn_wmma_f32_16x16x32_f16(false, a, false, b3, (short)0, c3, false, false);
    }

    // epilogue: bias add + f16 store.  C layout: M = j + 8*hi, N = lane&15
#pragma unroll
    for (int tix = 0; tix < 4; ++tix) {
        int n = n0 + tix * 16;
        float bn = bias[n];
        v8f c = (tix == 0) ? c0 : (tix == 1) ? c1 : (tix == 2) ? c2 : c3;
#pragma unroll
        for (int j = 0; j < 8; ++j) {
            int mo = mt * 16 + j + 8 * hi;
            xproj[(size_t)mo * KK + n] = (_Float16)(c[j] + bn);
        }
    }
}

// ---------- 4) persistent bidirectional RNN scan ----------
// grid = 2 (direction), block = 1024 (32 waves). Hidden state (G=24 rows,
// padded to 32, f16) double-buffered in LDS. Per step: H_new = tanh(xproj_t + H @ Wh).
// Wave w owns N-tile w (16 columns); B-fragment reused across both M-tiles.
__global__ void __launch_bounds__(1024)
eqr_rnn_kernel(const _Float16* __restrict__ xproj,   // (T*G, K) f16
               const _Float16* __restrict__ WhT,     // (K, K) transposed: [n][k]
               float* __restrict__ out) {            // hidden_all (T,G,2K) ++ ht (G,2K)
    __shared__ _Float16 Hbuf[2][32 * KK];            // 64 KB of the 320 KB WGP LDS

    const int dir  = blockIdx.x;                     // 0 = fwd, 1 = bwd
    const int tid  = threadIdx.x;
    const int lane = tid & 31;
    const int wave = tid >> 5;                       // 0..31
    const int hi   = lane >> 4;
    const int nl   = lane & 15;

    // zero both state buffers (covers the 8 pad rows, which must stay 0)
    for (int i = tid; i < 32 * KK; i += 1024) {
        Hbuf[0][i] = (_Float16)0.f;
        Hbuf[1][i] = (_Float16)0.f;
    }
    __syncthreads();

    const int n = wave * 16 + nl;                    // this wave's output column
    const _Float16* bcol = WhT + (size_t)n * KK + hi * 16;
    float* outHid = out;
    float* outHt  = out + (size_t)TT * GG * (2 * KK);

    int p = 0;
    for (int i = 0; i < TT; ++i) {
        // backward direction: rev = cat(xproj[:-1][::-1], xproj[-1:]) — same map for in & out
        const int ts = (dir == 0) ? i : ((i < TT - 1) ? (TT - 2 - i) : (TT - 1));

        // C init from xproj (f16 -> f32)
        v8f c0 = {}, c1 = {};
#pragma unroll
        for (int j = 0; j < 8; ++j) {
            int g0 = j + 8 * hi;                     // 0..15, always valid
            c0[j] = (float)xproj[((size_t)ts * GG + g0) * KK + n];
            int g1 = 16 + j + 8 * hi;                // 16..31, valid iff < 24
            c1[j] = (g1 < GG) ? (float)xproj[((size_t)ts * GG + g1) * KK + n] : 0.f;
        }

        const _Float16* Hc = &Hbuf[p][0];
        for (int k0 = 0; k0 < KK; k0 += 32) {
            h8 b0 = *(const h8*)(bcol + k0);
            h8 b1 = *(const h8*)(bcol + k0 + 8);
            v16h bf = SHUF16(b0, b1);                // one B-frag, two WMMAs

            const _Float16* ar0 = Hc + (size_t)nl * KK;
            h8 x0 = *(const h8*)(ar0 + k0 + hi * 16);
            h8 x1 = *(const h8*)(ar0 + k0 + hi * 16 + 8);
            v16h a0 = SHUF16(x0, x1);

            const _Float16* ar1 = Hc + (size_t)(16 + nl) * KK;
            h8 y0 = *(const h8*)(ar1 + k0 + hi * 16);
            h8 y1 = *(const h8*)(ar1 + k0 + hi * 16 + 8);
            v16h a1 = SHUF16(y0, y1);

            c0 = __builtin_amdgcn_wmma_f32_16x16x32_f16(false, a0, false, bf,
                                                        (short)0, c0, false, false);
            c1 = __builtin_amdgcn_wmma_f32_16x16x32_f16(false, a1, false, bf,
                                                        (short)0, c1, false, false);
        }

        _Float16* Hn = &Hbuf[1 - p][0];
#pragma unroll
        for (int j = 0; j < 8; ++j) {
            int g0 = j + 8 * hi;
            float hv0 = tanhf(c0[j]);
            Hn[(size_t)g0 * KK + n] = (_Float16)hv0;
            outHid[((size_t)ts * GG + g0) * (2 * KK) + (size_t)dir * KK + n] = hv0;
            if (i == TT - 1)
                outHt[(size_t)g0 * (2 * KK) + dir * KK + n] = hv0;

            int g1 = 16 + j + 8 * hi;
            if (g1 < GG) {
                float hv1 = tanhf(c1[j]);
                Hn[(size_t)g1 * KK + n] = (_Float16)hv1;
                outHid[((size_t)ts * GG + g1) * (2 * KK) + (size_t)dir * KK + n] = hv1;
                if (i == TT - 1)
                    outHt[(size_t)g1 * (2 * KK) + dir * KK + n] = hv1;
            }
        }
        __syncthreads();                             // H_new visible before next step
        p ^= 1;
    }
}

// ---------- workspace layout (bytes) ----------
#define OFF_TOK   0u
#define OFF_WE16  (8192u)                               // tok: 2048*4
#define OFF_WXT   (OFF_WE16 + (unsigned)(VV * KK * 2))  // We16: 2 MiB
#define OFF_WHT   (OFF_WXT + (unsigned)(KK * KK * 2))   // WxT: 512 KiB
#define OFF_XPROJ (OFF_WHT + (unsigned)(KK * KK * 2))   // WhT: 512 KiB; xproj: 48 MiB

extern "C" void kernel_launch(void* const* d_in, const int* in_sizes, int n_in,
                              void* d_out, int out_size, void* d_ws, size_t ws_size,
                              hipStream_t stream) {
    const float* seq   = (const float*)d_in[0];
    const int*   perms = (const int*)d_in[1];
    const float* W_e   = (const float*)d_in[2];
    const float* Wx    = (const float*)d_in[3];
    const float* Wh    = (const float*)d_in[4];
    const float* b     = (const float*)d_in[5];

    char* ws = (char*)d_ws;
    int*      tok    = (int*)(ws + OFF_TOK);
    _Float16* We16   = (_Float16*)(ws + OFF_WE16);
    _Float16* WxT    = (_Float16*)(ws + OFF_WXT);
    _Float16* WhT    = (_Float16*)(ws + OFF_WHT);
    _Float16* xproj  = (_Float16*)(ws + OFF_XPROJ);

    eqr_tok_kernel<<<TT, 256, 0, stream>>>(seq, tok);
    eqr_cvt_kernel<<<(VV * KK + 255) / 256, 256, 0, stream>>>(W_e, We16, VV, KK, 0);
    eqr_cvt_kernel<<<(KK * KK + 255) / 256, 256, 0, stream>>>(Wx, WxT, KK, KK, 1);
    eqr_cvt_kernel<<<(KK * KK + 255) / 256, 256, 0, stream>>>(Wh, WhT, KK, KK, 1);

    // (T*G/16) M-tiles x 8 N-groups (of 4 tiles), one 32-thread wave per strip
    eqr_xproj_kernel<<<(TT * GG / 16) * 8, 32, 0, stream>>>(
        We16, WxT, perms, tok, b, xproj);

    // 2 persistent workgroups: one per direction
    eqr_rnn_kernel<<<2, 1024, 0, stream>>>(xproj, WhT, (float*)d_out);
}
